// VectorQuantizer_9887014715904
// MI455X (gfx1250) — matrix operand order repaired and checked by reading, not compiled, not run on hardware
//
#include <hip/hip_runtime.h>
#include <hip/hip_bf16.h>
#include <math.h>

typedef __attribute__((ext_vector_type(16))) _Float16 v16h;
typedef __attribute__((ext_vector_type(8)))  float    v8f;

#define BATCH 256
#define LQ    1024
#define DQ    128
#define KQ    512
#define ROWS  (BATCH*LQ)   // 262144
#define DH    64           // D/2

// ---- workspace layout (bytes) ----
#define WS_PACKB   0ll                       // emb^T B-frags: 65536 halfs = 131072 B
#define WS_PACKW2  (WS_PACKB  + 131072ll)    // w2 B-frags:     8192 halfs =  16384 B
#define WS_E2      (WS_PACKW2 + 16384ll)     // 512 f32
#define WS_COUNTS  (WS_E2     + 2048ll)      // 512 f32
#define WS_CLSACC  (WS_COUNTS + 2048ll)      // 2560 f32
#define WS_SSE     (WS_CLSACC + 10240ll)     // 1 f32

// ---- output layout (floats, reference return order) ----
#define OUT_QST   1ll
#define OUT_PERP  (1ll + (long long)ROWS*DQ)
#define OUT_ENC   (2ll + (long long)ROWS*DQ)
#define OUT_CLS   (OUT_ENC + (long long)ROWS*KQ)

// ---------------------------------------------------------------- init
__global__ void vq_init(float* __restrict__ counts, float* __restrict__ clsacc,
                        float* __restrict__ sse) {
  int t = blockIdx.x * 256 + threadIdx.x;
  if (t < KQ) counts[t] = 0.0f;
  if (t < BATCH * 10) clsacc[t] = 0.0f;
  if (t == 0) *sse = 0.0f;
}

// --------------------------------------------- pack emb^T into WMMA B frags
// B frags as packed here (same k order as the A loads): col n = lane&15, K = (lane>>4)*16+E
__global__ void vq_pack_emb(const float* __restrict__ emb,
                            _Float16* __restrict__ packB) {
  int tid  = blockIdx.x * 256 + threadIdx.x;   // 65536
  int E    = tid & 15;
  int lane = (tid >> 4) & 31;
  int c    = (tid >> 9) & 3;
  int nt   = tid >> 11;
  int n    = nt * 16 + (lane & 15);
  int d    = c * 32 + (lane >> 4) * 16 + E;
  packB[tid] = (_Float16)emb[n * DQ + d];
}

// --------------------------------------------- pack w2 into WMMA B frags
// B[K=j][N=d]; tile dt (0..7) over d, kchunk c (0..1) over j
__global__ void vq_pack_w2(const float* __restrict__ w2,
                           _Float16* __restrict__ packW2) {
  int tid  = blockIdx.x * 256 + threadIdx.x;   // 8192
  int E    = tid & 15;
  int lane = (tid >> 4) & 31;
  int c    = (tid >> 9) & 1;
  int dt   = tid >> 10;
  int n    = dt * 16 + (lane & 15);
  int j    = c * 32 + (lane >> 4) * 16 + E;
  packW2[tid] = (_Float16)w2[j * DQ + n];
}

__global__ void vq_e2(const float* __restrict__ emb, float* __restrict__ e2) {
  int k = blockIdx.x * 256 + threadIdx.x;
  if (k >= KQ) return;
  float s = 0.0f;
  #pragma unroll 16
  for (int d = 0; d < DQ; ++d) { float v = emb[k * DQ + d]; s += v * v; }
  e2[k] = s;
}

// ------------------------------------------------------- fully fused core
// Per block: 16 rows. Encoder (2 WMMA/wave) -> distances (16 WMMA/wave)
// -> argmin -> q gather / one-hot / SSE / classifier partials.
__global__ __launch_bounds__(256) void vq_main(
    const float* __restrict__ x,   const float* __restrict__ emb,
    const float* __restrict__ w1,  const float* __restrict__ b1,
    const float* __restrict__ b2,
    const _Float16* __restrict__ packB, const _Float16* __restrict__ packW2,
    const float* __restrict__ e2,  const float* __restrict__ wc,
    float* __restrict__ counts, float* __restrict__ sse,
    float* __restrict__ clsacc, float* __restrict__ out) {
  __shared__ float    xs[16];
  __shared__ _Float16 hs[16 * DH];    // 2 KB
  __shared__ float    b2s[DQ];
  __shared__ float    zt[16 * DQ];    // 8 KB  f32 (loss)
  __shared__ _Float16 zt16[16 * DQ];  // 4 KB  f16 (A-matrix source)
  __shared__ float    sc[16 * KQ];    // 32 KB distances
  __shared__ float    e2s[KQ];        // 2 KB
  __shared__ float    bm[256];
  __shared__ int      bi[256];
  __shared__ int      rowSel[16];
  __shared__ float    ssum;
  __shared__ float    clsw[10];

  int t = threadIdx.x;
  long long r0 = (long long)blockIdx.x * 16;

  if (t < 16)  xs[t]  = x[r0 + t];
  if (t < DQ)  b2s[t] = b2[t];
  e2s[t]       = e2[t];
  e2s[t + 256] = e2[t + 256];
  if (t == 0)  ssum = 0.0f;
  if (t < 10)  clsw[t] = 0.0f;
  __syncthreads();

  // h = relu(x*w1+b1)  [16 x 64] f16
  #pragma unroll
  for (int i = 0; i < 4; ++i) {
    int idx = t * 4 + i;               // 0..1023
    int hm = idx >> 6, hj = idx & 63;
    float hv = xs[hm] * w1[hj] + b1[hj];
    hs[idx] = (_Float16)(hv > 0.0f ? hv : 0.0f);
  }
  __syncthreads();

  int w    = t >> 5;
  int lane = t & 31;
  int m    = lane & 15;
  int grp  = lane >> 4;

  // ---- encoder WMMA: wave w produces z columns d0 = w*16 .. +15
  {
    v8f zacc{};
    #pragma unroll
    for (int c = 0; c < 2; ++c) {
      v16h a;
      #pragma unroll
      for (int E = 0; E < 16; ++E) {
        int k = grp * 16 + E;
        a[E] = hs[m * DH + c * 32 + k];
      }
      const v16h b = *(const v16h*)(packW2 + (((w * 2 + c) * 32 + lane) << 4));
      zacc = __builtin_amdgcn_wmma_f32_16x16x32_f16(
          false, a, false, b, (short)0, zacc, false, false);
    }
    int d = w * 16 + m;
    #pragma unroll
    for (int i = 0; i < 8; ++i) {
      int row = i + 8 * grp;           // C layout: VGPR i -> M=i / M=i+8
      float zv = zacc[i] + b2s[d];
      zt[row * DQ + d]   = zv;
      zt16[row * DQ + d] = (_Float16)zv;
    }
  }
  __syncthreads();

  // ---- distance WMMA: wave w covers 4 column tiles of the 512 codes
  v8f acc[4] = {v8f{}, v8f{}, v8f{}, v8f{}};
  #pragma unroll
  for (int c = 0; c < 4; ++c) {
    v16h a;
    #pragma unroll
    for (int E = 0; E < 16; ++E) {
      int k = grp * 16 + E;
      a[E] = zt16[m * DQ + c * 32 + k];
    }
    #pragma unroll
    for (int tt = 0; tt < 4; ++tt) {
      int nt = w * 4 + tt;
      const v16h b = *(const v16h*)(packB + ((((long long)nt * 4 + c) * 32 + lane) << 4));
      acc[tt] = __builtin_amdgcn_wmma_f32_16x16x32_f16(
          false, a, false, b, (short)0, acc[tt], false, false);
    }
  }
  #pragma unroll
  for (int tt = 0; tt < 4; ++tt) {
    int col  = (w * 4 + tt) * 16 + m;
    float ev = e2s[col];
    #pragma unroll
    for (int i = 0; i < 8; ++i) {
      int row = i + 8 * grp;
      sc[row * KQ + col] = ev - 2.0f * acc[tt][i];
    }
  }
  __syncthreads();

  // ---- argmin (first-index tie-break, like jnp.argmin)
  int row = t >> 4, sub = t & 15;
  float best = 3.4e38f; int bidx = 0;
  for (int j = sub; j < KQ; j += 16) {
    float v = sc[row * KQ + j];
    if (v < best) { best = v; bidx = j; }
  }
  bm[t] = best; bi[t] = bidx;
  __syncthreads();
  if (sub == 0) {
    float bb = bm[row * 16]; int ii = bi[row * 16];
    #pragma unroll
    for (int j = 1; j < 16; ++j) {
      float v = bm[row * 16 + j]; int iv = bi[row * 16 + j];
      if (v < bb || (v == bb && iv < ii)) { bb = v; ii = iv; }
    }
    rowSel[row] = ii;
    atomicAdd(&counts[ii], 1.0f);
  }
  __syncthreads();

  int sel = rowSel[row];
  long long r = r0 + row;

  // ---- q_st gather + SSE partial (keep q values for classifier)
  float qv8[8];
  float part = 0.0f;
  #pragma unroll
  for (int i = 0; i < 8; ++i) {
    int d = sub * 8 + i;
    float qv = emb[sel * DQ + d];
    qv8[i] = qv;
    float dv = qv - zt[row * DQ + d];
    part += dv * dv;
    __builtin_nontemporal_store(qv, &out[OUT_QST + r * DQ + d]);
  }
  atomicAdd(&ssum, part);

  // ---- one-hot enc row (537 MB stream: non-temporal)
  for (int j = sub; j < KQ; j += 16)
    __builtin_nontemporal_store((j == sel) ? 1.0f : 0.0f,
                                &out[OUT_ENC + r * KQ + j]);

  // ---- classifier partials: this thread owns q[b, l*128 + sub*8 .. +7]
  int l = (int)(r & (LQ - 1));
  float cp[10];
  #pragma unroll
  for (int o = 0; o < 10; ++o) cp[o] = 0.0f;
  #pragma unroll
  for (int i = 0; i < 8; ++i) {
    long long wrow = (long long)(l * DQ + sub * 8 + i) * 10;
    #pragma unroll
    for (int o = 0; o < 10; ++o) cp[o] += qv8[i] * wc[wrow + o];
  }
  #pragma unroll
  for (int o = 0; o < 10; ++o) {
    float v = cp[o];
    #pragma unroll
    for (int off = 16; off > 0; off >>= 1) v += __shfl_down(v, off, 32);
    if (lane == 0) atomicAdd(&clsw[o], v);
  }
  __syncthreads();
  if (t < 10) atomicAdd(&clsacc[(int)(r0 >> 10) * 10 + t], clsw[t]);
  if (t == 0) atomicAdd(sse, ssum);
}

// ------------------------------------------------------------- classifier fin
__global__ void vq_cls_fin(const float* __restrict__ clsacc,
                           const float* __restrict__ bc,
                           float* __restrict__ out_cls) {
  int t = blockIdx.x * 256 + threadIdx.x;
  if (t >= BATCH * 10) return;
  float v = clsacc[t] + bc[t % 10];
  out_cls[t] = 1.0f / (1.0f + __expf(-v));
}

// ------------------------------------------------------------- finalize
__global__ __launch_bounds__(512) void vq_finalize(
    const float* __restrict__ counts, const float* __restrict__ sse,
    float* __restrict__ out) {
  __shared__ float red[512];
  int t = threadIdx.x;
  float p = counts[t] * (1.0f / (float)ROWS);
  red[t] = p * __logf(p + 1e-10f);
  __syncthreads();
  for (int s = 256; s > 0; s >>= 1) {
    if (t < s) red[t] += red[t + s];
    __syncthreads();
  }
  if (t == 0) {
    out[OUT_PERP] = __expf(-red[0]);
    // q_latent + COMMIT*e_latent = 1.25 * mean((q-z)^2)
    out[0] = 1.25f * sse[0] / (float)((long long)ROWS * DQ);
  }
}

// ---------------------------------------------------------------- launch
extern "C" void kernel_launch(void* const* d_in, const int* in_sizes, int n_in,
                              void* d_out, int out_size, void* d_ws, size_t ws_size,
                              hipStream_t stream) {
  const float* x   = (const float*)d_in[0];
  const float* emb = (const float*)d_in[1];
  const float* w1  = (const float*)d_in[2];
  const float* b1  = (const float*)d_in[3];
  const float* w2  = (const float*)d_in[4];
  const float* b2  = (const float*)d_in[5];
  const float* wc  = (const float*)d_in[6];
  const float* bc  = (const float*)d_in[7];
  float* out = (float*)d_out;

  char* ws = (char*)d_ws;
  _Float16* packB  = (_Float16*)(ws + WS_PACKB);
  _Float16* packW2 = (_Float16*)(ws + WS_PACKW2);
  float*    e2     = (float*)(ws + WS_E2);
  float*    counts = (float*)(ws + WS_COUNTS);
  float*    clsacc = (float*)(ws + WS_CLSACC);
  float*    ssePtr = (float*)(ws + WS_SSE);

  vq_init    <<<13,       256, 0, stream>>>(counts, clsacc, ssePtr);
  vq_pack_emb<<<256,      256, 0, stream>>>(emb, packB);
  vq_pack_w2 <<<32,       256, 0, stream>>>(w2, packW2);
  vq_e2      <<<2,        256, 0, stream>>>(emb, e2);
  vq_main    <<<ROWS/16,  256, 0, stream>>>(x, emb, w1, b1, b2, packB, packW2,
                                            e2, wc, counts, ssePtr, clsacc, out);
  vq_cls_fin <<<10,       256, 0, stream>>>(clsacc, bc, out + OUT_CLS);
  vq_finalize<<<1,        512, 0, stream>>>(counts, ssePtr, out);
}
